// HigherOrderGINLayer_36369783062756
// MI455X (gfx1250) — matrix operand order repaired and checked by reading, not compiled, not run on hardware
//
#include <hip/hip_runtime.h>

typedef __attribute__((ext_vector_type(16))) _Float16 v16h;
typedef __attribute__((ext_vector_type(8)))  _Float16 v8h;
typedef __attribute__((ext_vector_type(4)))  _Float16 v4h;
typedef __attribute__((ext_vector_type(8)))  float    v8f;

#define NM   50000   // nodes
#define NE   800000  // edges per adjacency
#define DF   256     // feature dim (DIN == DOUT == 256)
#define PF   128     // attention projection dim
#define KK   256     // every GEMM in this net has K = 256

// ---------------------------------------------------------------- utilities

__global__ void k_zero(float* __restrict__ p, int n) {
  int i = blockIdx.x * blockDim.x + threadIdx.x;
  if (i < n) p[i] = 0.0f;
}

// Pack f32 weight W[K=256][N] into WMMA B-fragment order, f16:
// P[((nt*8 + kb)*32 + lane)*16 + e] = W[kb*32 + (lane>>4)*16 + e][nt*16 + (lane&15)]
// so in the GEMM each lane loads its whole fragment as 32 contiguous bytes.
__global__ void k_pack_w(const float* __restrict__ W, _Float16* __restrict__ P, int N) {
  int i = blockIdx.x * blockDim.x + threadIdx.x;
  if (i >= N * KK) return;
  int e    = i & 15;
  int lane = (i >> 4) & 31;
  int kb   = (i >> 9) & 7;
  int nt   = i >> 12;
  int krow = kb * 32 + (lane >> 4) * 16 + e;
  int col  = nt * 16 + (lane & 15);
  P[i] = (_Float16)W[(size_t)krow * N + col];
}

// ------------------------------------------------------- WMMA GEMM (fused)
// C[M,N] = act(A[M,256] @ W[256,N] + bias[N]) (+ resid[M,N])
// block = 256 threads (8 wave32); block tile 64x64; wave tile 16x32
// (2 accumulators, shared A fragment); full A panel staged in LDS once.
__global__ __launch_bounds__(256) void k_gemm(
    const float* __restrict__ A, const _Float16* __restrict__ Wp,
    const float* __restrict__ bias, const float* __restrict__ resid,
    float* __restrict__ C, int M, int N, int relu)
{
  __shared__ __align__(16) _Float16 Ash[64 * 264];   // 64 rows x 256 halves, stride 264

  const int tid  = threadIdx.x;
  const int wave = tid >> 5;
  const int lane = tid & 31;
  const int m0 = blockIdx.x * 64;
  const int n0 = blockIdx.y * 64;
  const int wm = (wave & 3) * 16;    // wave row offset in tile
  const int wn = (wave >> 2) * 32;   // wave col offset in tile (2x16 wide)
  const int hi = lane >> 4;
  const int lr = lane & 15;

  // ---- stage full 64x256 A panel (f32 -> f16), coalesced float4 loads,
  //      branch-free OOB handling via clamp + mask
  #pragma unroll
  for (int i = 0; i < 16; ++i) {
    int q = tid + i * 256;           // quad index, 4096 total
    int r = q >> 6;                  // row 0..63
    int c = (q & 63) * 4;            // k column 0..252
    int gr  = m0 + r;
    int grc = (gr < M) ? gr : (M - 1);
    float msk = (gr < M) ? 1.0f : 0.0f;
    float4 v = *(const float4*)(A + (size_t)grc * KK + c);
    v4h h;
    h[0] = (_Float16)(v.x * msk);
    h[1] = (_Float16)(v.y * msk);
    h[2] = (_Float16)(v.z * msk);
    h[3] = (_Float16)(v.w * msk);
    *(v4h*)&Ash[r * 264 + c] = h;    // ds_store_b64
  }
  __syncthreads();

  // packed-B base for this wave's first 16-col tile; second tile is +4096 halves
  const _Float16* bp = Wp + ((size_t)(n0 + wn) >> 4) * (KK * 16) + lane * 16;
  const int arow = (wm + lr) * 264;

  v8f acc[2] = {{}, {}};

  #pragma unroll
  for (int kb = 0; kb < 8; ++kb) {
    // A fragment: two contiguous 16B chunks per lane -> 2x ds_load_b128
    const int ao = arow + kb * 32 + hi * 16;
    v8h a0 = *(const v8h*)&Ash[ao];
    v8h a1 = *(const v8h*)&Ash[ao + 8];
    v16h a = __builtin_shufflevector(a0, a1, 0,1,2,3,4,5,6,7,8,9,10,11,12,13,14,15);

    // B fragments: 32 contiguous bytes per lane per tile -> global_load_b128 x2
    const _Float16* b0p = bp + kb * 512;
    v8h b0l = *(const v8h*)(b0p);
    v8h b0h = *(const v8h*)(b0p + 8);
    v16h b0 = __builtin_shufflevector(b0l, b0h, 0,1,2,3,4,5,6,7,8,9,10,11,12,13,14,15);
    const _Float16* b1p = b0p + 4096;
    v8h b1l = *(const v8h*)(b1p);
    v8h b1h = *(const v8h*)(b1p + 8);
    v16h b1 = __builtin_shufflevector(b1l, b1h, 0,1,2,3,4,5,6,7,8,9,10,11,12,13,14,15);

    acc[0] = __builtin_amdgcn_wmma_f32_16x16x32_f16(
        false, a, false, b0, (short)0, acc[0], false, false);
    acc[1] = __builtin_amdgcn_wmma_f32_16x16x32_f16(
        false, a, false, b1, (short)0, acc[1], false, false);
  }

  // ---- epilogue: bias (+relu) (+resid), C/D layout row = wm + 8*hi + j, col = 16-tile
  #pragma unroll
  for (int t = 0; t < 2; ++t) {
    const int col = n0 + wn + t * 16 + lr;
    const float bv = bias[col];
    #pragma unroll
    for (int j = 0; j < 8; ++j) {
      int row = m0 + wm + hi * 8 + j;
      if (row < M) {
        float v = acc[t][j] + bv;
        if (relu) v = fmaxf(v, 0.0f);
        if (resid) v += resid[(size_t)row * N + col];
        C[(size_t)row * N + col] = v;
      }
    }
  }
}

// ------------------------------------------------------- SpMM (wave/edge)
// y[row[e]] += val[e] * x[col[e]] ; 32 lanes x 8 floats = one 1KB row/edge
__global__ void k_spmm(const int* __restrict__ rows, const int* __restrict__ cols,
                       const float* __restrict__ vals, const float* __restrict__ x,
                       float* __restrict__ y, int E)
{
  int gid  = blockIdx.x * blockDim.x + threadIdx.x;
  int e    = gid >> 5;
  int lane = gid & 31;
  if (e >= E) return;
  int r = rows[e];
  int c = cols[e];
  float v = vals[e];
  const float4* xr = (const float4*)(x + (size_t)c * DF);
  float* yr = y + (size_t)r * DF + lane * 8;
  float4 f0 = xr[lane * 2 + 0];
  float4 f1 = xr[lane * 2 + 1];
  atomicAdd(yr + 0, v * f0.x);
  atomicAdd(yr + 1, v * f0.y);
  atomicAdd(yr + 2, v * f0.z);
  atomicAdd(yr + 3, v * f0.w);
  atomicAdd(yr + 4, v * f1.x);
  atomicAdd(yr + 5, v * f1.y);
  atomicAdd(yr + 6, v * f1.z);
  atomicAdd(yr + 7, v * f1.w);
}

// -------------------------------------------- attention scores (wave/node)
__global__ void k_rowdot(const float* __restrict__ q, const float* __restrict__ keys,
                         float* __restrict__ sc, int k)
{
  int gid  = blockIdx.x * blockDim.x + threadIdx.x;
  int n    = gid >> 5;
  int lane = gid & 31;
  if (n >= NM) return;
  const float4* qa = (const float4*)(q    + (size_t)n * PF);
  const float4* ka = (const float4*)(keys + (size_t)n * PF);
  float4 qv = qa[lane], kv = ka[lane];
  float s = qv.x * kv.x + qv.y * kv.y + qv.z * kv.z + qv.w * kv.w;
  s += __shfl_xor(s, 16);
  s += __shfl_xor(s, 8);
  s += __shfl_xor(s, 4);
  s += __shfl_xor(s, 2);
  s += __shfl_xor(s, 1);
  if (lane == 0) sc[(size_t)n * 5 + k] = s;
}

// ------------------------------------- softmax + weighted combine (wave/node)
__global__ void k_attn(const float* __restrict__ sc,
                       const float* __restrict__ f0, const float* __restrict__ f1,
                       const float* __restrict__ f2, const float* __restrict__ f3,
                       const float* __restrict__ f4, float* __restrict__ out)
{
  int gid  = blockIdx.x * blockDim.x + threadIdx.x;
  int n    = gid >> 5;
  int lane = gid & 31;
  if (n >= NM) return;
  float s0 = sc[(size_t)n * 5 + 0], s1 = sc[(size_t)n * 5 + 1],
        s2 = sc[(size_t)n * 5 + 2], s3 = sc[(size_t)n * 5 + 3],
        s4 = sc[(size_t)n * 5 + 4];
  float m  = fmaxf(fmaxf(fmaxf(s0, s1), fmaxf(s2, s3)), s4);
  float w[5];
  w[0] = __expf(s0 - m); w[1] = __expf(s1 - m); w[2] = __expf(s2 - m);
  w[3] = __expf(s3 - m); w[4] = __expf(s4 - m);
  float inv = 1.0f / (w[0] + w[1] + w[2] + w[3] + w[4]);
  #pragma unroll
  for (int k = 0; k < 5; ++k) w[k] *= inv;

  size_t base = (size_t)n * DF + lane * 8;
  const float* fs[5] = {f0 + base, f1 + base, f2 + base, f3 + base, f4 + base};
  float4 r0 = {0.f, 0.f, 0.f, 0.f}, r1 = {0.f, 0.f, 0.f, 0.f};
  #pragma unroll
  for (int k = 0; k < 5; ++k) {
    const float4* p = (const float4*)fs[k];
    float4 x0 = p[0], x1 = p[1];
    r0.x += w[k] * x0.x; r0.y += w[k] * x0.y; r0.z += w[k] * x0.z; r0.w += w[k] * x0.w;
    r1.x += w[k] * x1.x; r1.y += w[k] * x1.y; r1.z += w[k] * x1.z; r1.w += w[k] * x1.w;
  }
  float4* o = (float4*)(out + base);
  o[0] = r0; o[1] = r1;
}

// ---------------------------------------------------------------- launcher

extern "C" void kernel_launch(void* const* d_in, const int* in_sizes, int n_in,
                              void* d_out, int out_size, void* d_ws, size_t ws_size,
                              hipStream_t stream)
{
  (void)in_sizes; (void)n_in; (void)out_size; (void)ws_size;

  const float* features = (const float*)d_in[0];
  const float* edgefeat = (const float*)d_in[1];
  const int*   adjI[4]  = {(const int*)d_in[2], (const int*)d_in[4],
                           (const int*)d_in[6], (const int*)d_in[8]};
  const float* adjV[4]  = {(const float*)d_in[3], (const float*)d_in[5],
                           (const float*)d_in[7], (const float*)d_in[9]};
  const float* ftB = (const float*)d_in[11];
  const float* qW  = (const float*)d_in[36];
  const float* qB  = (const float*)d_in[37];
  const float* kW  = (const float*)d_in[38];
  const float* kB  = (const float*)d_in[39];

  char* ws = (char*)d_ws;
  size_t off = 0;
  auto take = [&](size_t bytes) -> void* {
    void* p = ws + off;
    off = (off + bytes + 255) & ~(size_t)255;
    return p;
  };

  // packed f16 weights: ft, fo1, fo2, em1, em2, so1, so2, to1, to2, qo1, qo2, m1, m2
  _Float16* hW[13];
  for (int i = 0; i < 13; ++i) hW[i] = (_Float16*)take((size_t)DF * DF * sizeof(_Float16));
  _Float16* hQ = (_Float16*)take((size_t)KK * PF * sizeof(_Float16));
  _Float16* hK = (_Float16*)take((size_t)KK * PF * sizeof(_Float16));

  float* feats = (float*)take((size_t)NM * DF * 4);
  float* fo    = (float*)take((size_t)NM * DF * 4);
  float* so    = (float*)take((size_t)NM * DF * 4);
  float* to    = (float*)take((size_t)NM * DF * 4);
  float* qo    = (float*)take((size_t)NM * DF * 4);
  float* tA    = (float*)take((size_t)NM * DF * 4);
  float* tB    = (float*)take((size_t)NM * DF * 4);
  float* qbuf  = (float*)take((size_t)NM * PF * 4);
  float* sc    = (float*)take((size_t)NM * 5 * 4);

  // pack all weight matrices into WMMA fragment order (f16)
  const float* srcW[13] = {
      (const float*)d_in[10],
      (const float*)d_in[12], (const float*)d_in[14],
      (const float*)d_in[16], (const float*)d_in[18],
      (const float*)d_in[20], (const float*)d_in[22],
      (const float*)d_in[24], (const float*)d_in[26],
      (const float*)d_in[28], (const float*)d_in[30],
      (const float*)d_in[32], (const float*)d_in[34]};
  for (int i = 0; i < 13; ++i)
    k_pack_w<<<(DF * KK + 255) / 256, 256, 0, stream>>>(srcW[i], hW[i], DF);
  k_pack_w<<<(PF * KK + 255) / 256, 256, 0, stream>>>(qW, hQ, PF);
  k_pack_w<<<(PF * KK + 255) / 256, 256, 0, stream>>>(kW, hK, PF);

  auto gemm = [&](const float* Ap, const _Float16* Wp, const float* Bp,
                  const float* Rp, float* Cp, int Nn, int relu) {
    dim3 g((NM + 63) / 64, Nn / 64);
    k_gemm<<<g, 256, 0, stream>>>(Ap, Wp, Bp, Rp, Cp, NM, Nn, relu);
  };
  auto zero = [&](float* p, size_t n) {
    k_zero<<<((int)n + 255) / 256, 256, 0, stream>>>(p, (int)n);
  };
  auto spmm = [&](int a, const float* x, float* y) {
    dim3 g((NE * 32 + 255) / 256);
    k_spmm<<<g, 256, 0, stream>>>(adjI[a], adjI[a] + NE, adjV[a], x, y, NE);
  };

  // feats = features @ ft_W + ft_b
  gemm(features, hW[0], ftB, nullptr, feats, DF, 0);

  // first order: fo = MLP(spmm(adj1, feats)) + spmm(adj1, MLP(edge_features))
  zero(tA, (size_t)NM * DF);
  spmm(0, feats, tA);
  gemm(tA, hW[1], (const float*)d_in[13], nullptr, tB, DF, 1);
  gemm(tB, hW[2], (const float*)d_in[15], nullptr, fo, DF, 0);
  gemm(edgefeat, hW[3], (const float*)d_in[17], nullptr, tA, DF, 1);
  gemm(tA, hW[4], (const float*)d_in[19], nullptr, tB, DF, 0);
  spmm(0, tB, fo);   // accumulates into fo

  // higher orders: out = MLP(agg) + agg
  const int   oAdj[3] = {1, 2, 3};
  const int   oW1[3]  = {5, 7, 9},    oW2[3] = {6, 8, 10};
  const int   oB1[3]  = {21, 25, 29}, oB2[3] = {23, 27, 31};
  float*      oOut[3] = {so, to, qo};
  for (int i = 0; i < 3; ++i) {
    zero(tA, (size_t)NM * DF);
    spmm(oAdj[i], feats, tA);
    gemm(tA, hW[oW1[i]], (const float*)d_in[oB1[i]], nullptr, tB, DF, 1);
    gemm(tB, hW[oW2[i]], (const float*)d_in[oB2[i]], tA, oOut[i], DF, 0);
  }

  // attention: q, per-branch keys -> scores -> softmax -> combine
  gemm(feats, hQ, qB, nullptr, qbuf, PF, 0);
  const float* Fs[5] = {feats, fo, so, to, qo};
  dim3 gn((NM * 32 + 255) / 256);
  for (int k = 0; k < 5; ++k) {
    gemm(Fs[k], hK, kB, nullptr, tB, PF, 0);   // keys (NM x 128) fits in tB
    k_rowdot<<<gn, 256, 0, stream>>>(qbuf, tB, sc, k);
  }
  k_attn<<<gn, 256, 0, stream>>>(sc, feats, fo, so, to, qo, tA);

  // final MLP -> d_out
  gemm(tA, hW[11], (const float*)d_in[33], nullptr, tB, DF, 1);
  gemm(tB, hW[12], (const float*)d_in[35], nullptr, (float*)d_out, DF, 0);
}
